// FragC3_86122684220181
// MI455X (gfx1250) — matrix operand order repaired and checked by reading, not compiled, not run on hardware
//
#include <hip/hip_runtime.h>
#include <hip/hip_bf16.h>
#include <cstdint>

typedef _Float16 half_t;
typedef __attribute__((ext_vector_type(16))) _Float16 v16h;
typedef __attribute__((ext_vector_type(8)))  float    v8f;

#define B_   48
#define L_   512
#define D_   192
#define H_   6
#define DK_  32
#define HID_ 768
#define M_   (B_ * L_)                  // 24576 rows per stream
#define SCALE_ 0.17677669529663689f     // 1/sqrt(32)
#define LDS_T 40                        // LDS tile stride in halves (80 B, 16B-aligned)

// ---------------- wave32 helpers ----------------
__device__ __forceinline__ float wave_sum(float x) {
#pragma unroll
  for (int o = 16; o > 0; o >>= 1) x += __shfl_xor(x, o, 32);
  return x;
}

// ---------------- WMMA fragment loaders (16x16x32 f16, wave32) ----------------
// A 16x32 from row-major [M][K] tile. Per-lane data is two contiguous 8-half
// chunks: k in [8*half, 8*half+8) and [16+8*half, 16+8*half+8)  -> 2x b128.
__device__ __forceinline__ v16h load_a_frag_vec(const half_t* src, int ld) {
  const int lane = threadIdx.x & 31;
  const int m = lane & 15, hl = lane >> 4;
  const half_t* p = src + m * ld + 8 * hl;
  v16h r;
  *reinterpret_cast<uint4*>(&r)       = *reinterpret_cast<const uint4*>(p);
  *(reinterpret_cast<uint4*>(&r) + 1) = *reinterpret_cast<const uint4*>(p + 16);
  return r;
}
// B 32x16 from an [N][K] tile (i.e. B transposed in memory). Per-lane data is two
// contiguous 8-half chunks: k in [8*half, 8*half+8) and [16+8*half, 16+8*half+8)  -> 2x b128.
__device__ __forceinline__ v16h load_bt_frag_vec(const half_t* src, int ld) {
  const int lane = threadIdx.x & 31;
  const int n = lane & 15, hl = lane >> 4;
  const half_t* p = src + n * ld + 8 * hl;
  v16h r;
  *reinterpret_cast<uint4*>(&r)       = *reinterpret_cast<const uint4*>(p);
  *(reinterpret_cast<uint4*>(&r) + 1) = *reinterpret_cast<const uint4*>(p + 16);
  return r;
}

#define WMMA_F32_F16(a, b, c) \
  __builtin_amdgcn_wmma_f32_16x16x32_f16(false, (a), false, (b), (short)0, (c), false, false)

// ---------------- shared GEMM mainloop ----------------
// Block tile 128(M) x 64(N), BK=32, 256 threads = 8 waves.
// A tile: [128][32] row-major activations (2x b128 global loads/thread);
// B tile: [64][32] from [N][K] pre-transposed weights (1x b128/thread).
// Wave w: mpair = w>>1 -> rows mpair*32 (two 16-row tiles);
//         npair = w&1  -> cols npair*32 (two 16-col tiles).
// 4 WMMAs per k-step from 8 ds_load_b128 (2 LDS loads per WMMA).
__device__ __forceinline__ void gemm_mainloop(const half_t* __restrict__ Ag, int lda,
                                              const half_t* __restrict__ BTg, int ldbt,
                                              int K,
                                              half_t* sA, half_t* sBt,
                                              v8f (&acc)[2][2],
                                              int mpair, int npair) {
  const int t = threadIdx.x;
  const int row = t >> 2, col = (t & 3) << 3;  // 64 rows x 32 halves per pass
  for (int k0 = 0; k0 < K; k0 += 32) {
    *reinterpret_cast<uint4*>(sA + row * LDS_T + col) =
        *reinterpret_cast<const uint4*>(Ag + (size_t)row * lda + k0 + col);
    *reinterpret_cast<uint4*>(sA + (row + 64) * LDS_T + col) =
        *reinterpret_cast<const uint4*>(Ag + (size_t)(row + 64) * lda + k0 + col);
    *reinterpret_cast<uint4*>(sBt + row * LDS_T + col) =
        *reinterpret_cast<const uint4*>(BTg + (size_t)row * ldbt + k0 + col);
    __syncthreads();
    v16h a0 = load_a_frag_vec(sA + (mpair * 32) * LDS_T, LDS_T);
    v16h a1 = load_a_frag_vec(sA + (mpair * 32 + 16) * LDS_T, LDS_T);
    v16h b0 = load_bt_frag_vec(sBt + (npair * 32) * LDS_T, LDS_T);
    v16h b1 = load_bt_frag_vec(sBt + (npair * 32 + 16) * LDS_T, LDS_T);
    acc[0][0] = WMMA_F32_F16(a0, b0, acc[0][0]);
    acc[0][1] = WMMA_F32_F16(a0, b1, acc[0][1]);
    acc[1][0] = WMMA_F32_F16(a1, b0, acc[1][0]);
    acc[1][1] = WMMA_F32_F16(a1, b1, acc[1][1]);
    __syncthreads();
  }
}

// ---------------- weight fp32 -> f16, transposed to [N][K] ----------------
__global__ __launch_bounds__(256) void cvt_w_kernel(
    const float* __restrict__ wq, const float* __restrict__ wk,
    const float* __restrict__ wv, const float* __restrict__ wo,
    const float* __restrict__ w1, const float* __restrict__ w2,
    half_t* __restrict__ out) {
  const int S = D_ * D_;       // 36864
  const int SW = D_ * HID_;    // 147456
  int i = blockIdx.x * 256 + threadIdx.x;
  if (i >= 4 * S + 2 * SW) return;
  float v;
  if (i < 4 * S) {               // wq,wk,wv,wo : [D][D] -> [N][K]
    const int w = i / S, j = i % S;
    const int n = j / D_, k = j % D_;
    const float* src = (w == 0) ? wq : (w == 1) ? wk : (w == 2) ? wv : wo;
    v = src[k * D_ + n];
  } else if (i < 4 * S + SW) {   // w1 : [D][HID] -> [HID][D]
    const int j = i - 4 * S;
    const int n = j / D_, k = j % D_;
    v = w1[k * HID_ + n];
  } else {                       // w2 : [HID][D] -> [D][HID]
    const int j = i - 4 * S - SW;
    const int n = j / HID_, k = j % HID_;
    v = w2[k * D_ + n];
  }
  out[i] = (half_t)v;
}

// ---------------- input masked LN -> f16 ----------------
__global__ __launch_bounds__(256) void ln_in_kernel(
    const float* __restrict__ xa, const float* __restrict__ xb,
    const unsigned char* __restrict__ va, const unsigned char* __restrict__ vb,
    const float* __restrict__ ga, const float* __restrict__ ba,
    const float* __restrict__ gb, const float* __restrict__ bbp,
    half_t* __restrict__ outa, half_t* __restrict__ outb) {
  const int z = blockIdx.y;
  const float* x = z ? xb : xa;
  const unsigned char* vv = z ? vb : va;
  const float* g = z ? gb : ga;
  const float* be = z ? bbp : ba;
  half_t* out = z ? outb : outa;
  const int wid = threadIdx.x >> 5, lane = threadIdx.x & 31;
  const size_t row = (size_t)blockIdx.x * 8 + wid;
  const float* rp = x + row * D_;
  float xv[6]; float s = 0.f, ss = 0.f;
#pragma unroll
  for (int i = 0; i < 6; ++i) { xv[i] = rp[lane + 32 * i]; s += xv[i]; ss += xv[i] * xv[i]; }
  s = wave_sum(s); ss = wave_sum(ss);
  const float mu = s * (1.f / D_);
  const float rstd = rsqrtf(ss * (1.f / D_) - mu * mu + 1e-5f);
  const bool valid = vv[row] != 0;
  half_t* op = out + row * D_;
#pragma unroll
  for (int i = 0; i < 6; ++i) {
    const int c = lane + 32 * i;
    const float y = valid ? (xv[i] - mu) * rstd * g[c] + be[c] : xv[i];
    op[c] = (half_t)y;
  }
}

// ---------------- QKV projection ----------------
// Q/K stored [B,H,L,dk] (scale folded into Q); V stored transposed [B,H,dk,L]
// so the attention PV step gets contiguous B-fragments.
__global__ __launch_bounds__(256) void proj_kernel(
    const half_t* __restrict__ A16, const half_t* __restrict__ B16,
    const half_t* __restrict__ WQT, const half_t* __restrict__ WKT,
    const half_t* __restrict__ WVT,
    const float* __restrict__ bq, const float* __restrict__ bk,
    const float* __restrict__ bv,
    half_t* QA, half_t* KA, half_t* VAT, half_t* QB, half_t* KB, half_t* VBT) {
  __shared__ alignas(16) half_t sA[128 * LDS_T];
  __shared__ alignas(16) half_t sBt[64 * LDS_T];
  const int z = blockIdx.z;
  const int stream = z / 3, wsel = z % 3;
  const half_t* X = stream ? B16 : A16;
  const half_t* WT = (wsel == 0) ? WQT : (wsel == 1) ? WKT : WVT;
  const float* bias = (wsel == 0) ? bq : (wsel == 1) ? bk : bv;
  half_t* outs0[3] = {QA, KA, VAT};
  half_t* outs1[3] = {QB, KB, VBT};
  half_t* out = stream ? outs1[wsel] : outs0[wsel];
  const float sc = (wsel == 0) ? SCALE_ : 1.f;
  const int wid = threadIdx.x >> 5, lane = threadIdx.x & 31;
  const int mpair = wid >> 1, npair = wid & 1;
  const int n = lane & 15, hl = lane >> 4;
  v8f acc[2][2] = {};
  gemm_mainloop(X + (size_t)blockIdx.x * 128 * D_, D_,
                WT + (size_t)blockIdx.y * 64 * D_, D_, D_,
                sA, sBt, acc, mpair, npair);
  const int rowBase = blockIdx.x * 128 + mpair * 32 + hl * 8;
  const int colBase = blockIdx.y * 64 + npair * 32;
#pragma unroll
  for (int mi = 0; mi < 2; ++mi) {
#pragma unroll
    for (int ni = 0; ni < 2; ++ni) {
      const v8f& a = acc[mi][ni];
      const int col = colBase + ni * 16 + n;
      const float bi = bias[col];
      const int h = col >> 5, dk = col & 31;
#pragma unroll
      for (int r = 0; r < 8; ++r) {
        const int row = rowBase + mi * 16 + r;
        const int bb = row >> 9, l = row & (L_ - 1);
        const half_t val = (half_t)((a[r] + bi) * sc);
        if (wsel == 2)  // V transposed: [B,H,dk,L]
          out[(((size_t)bb * H_ + h) * DK_ + dk) * L_ + l] = val;
        else            // Q/K: [B,H,L,dk]
          out[(((size_t)bb * H_ + h) * L_ + l) * DK_ + dk] = val;
      }
    }
  }
}

// ---------------- flash attention (both directions) ----------------
__global__ __launch_bounds__(256) void attn_kernel(
    const half_t* __restrict__ QA, const half_t* __restrict__ KB,
    const half_t* __restrict__ VBT,
    const half_t* __restrict__ QB, const half_t* __restrict__ KA,
    const half_t* __restrict__ VAT,
    const unsigned char* __restrict__ valid_a,
    const unsigned char* __restrict__ valid_b,
    half_t* __restrict__ OA, half_t* __restrict__ OB) {
  __shared__ alignas(16) half_t sK[32 * LDS_T];   // [key][dk]  ([N][K] for QK^T)
  __shared__ alignas(16) half_t sVt[32 * LDS_T];  // [dk][key]  ([N][K] for PV)
  __shared__ alignas(16) half_t sP[8][16 * LDS_T];

  const int dir = blockIdx.y;
  const half_t* Q = dir ? QB : QA;
  const half_t* Km = dir ? KA : KB;
  const half_t* Vt = dir ? VAT : VBT;
  const unsigned char* vq = dir ? valid_b : valid_a;
  const unsigned char* vk = dir ? valid_a : valid_b;
  half_t* Om = dir ? OB : OA;

  const int bh = blockIdx.x >> 2;
  const int qchunk = blockIdx.x & 3;
  const int bb = bh / H_;
  const int h = bh % H_;
  const int wid = threadIdx.x >> 5;
  const int lane = threadIdx.x & 31;
  const int n = lane & 15, hl = lane >> 4;

  const size_t headoff = (size_t)bh * L_ * DK_;
  const half_t* Qh = Q + headoff;
  const half_t* Kh = Km + headoff;
  const half_t* Vh = Vt + headoff;  // [dk][L] within this head
  const unsigned char* vqp = vq + bb * L_;
  const unsigned char* vkp = vk + bb * L_;

  const int qbase = (qchunk * 8 + wid) * 16;

  // Q fragment (A layout) straight from global: 2x global_load_b128 per lane
  v16h qf;
  {
    const half_t* p = Qh + (size_t)(qbase + n) * DK_ + 8 * hl;
    *reinterpret_cast<uint4*>(&qf)       = *reinterpret_cast<const uint4*>(p);
    *(reinterpret_cast<uint4*>(&qf) + 1) = *reinterpret_cast<const uint4*>(p + 16);
  }

  bool vqb[8];
#pragma unroll
  for (int r = 0; r < 8; ++r) vqb[r] = vqp[qbase + hl * 8 + r] != 0;

  float m_run[8], l_run[8];
#pragma unroll
  for (int r = 0; r < 8; ++r) { m_run[r] = -1e30f; l_run[r] = 0.f; }
  v8f o0 = {}, o1 = {};

  for (int kt = 0; kt < L_ / 32; ++kt) {
    {  // cooperative tile loads: K rows = keys, V^T rows = dk; uint2 per thread
      const int row = threadIdx.x >> 3;
      const int ch = (threadIdx.x & 7) << 2;
      *reinterpret_cast<uint2*>(&sK[row * LDS_T + ch]) =
          *reinterpret_cast<const uint2*>(Kh + (size_t)(kt * 32 + row) * DK_ + ch);
      *reinterpret_cast<uint2*>(&sVt[row * LDS_T + ch]) =
          *reinterpret_cast<const uint2*>(Vh + (size_t)row * L_ + kt * 32 + ch);
    }
    __syncthreads();

    v8f s0 = {}, s1 = {};
    {
      v16h bk0 = load_bt_frag_vec(sK, LDS_T);
      v16h bk1 = load_bt_frag_vec(sK + 16 * LDS_T, LDS_T);
      s0 = WMMA_F32_F16(qf, bk0, s0);
      s1 = WMMA_F32_F16(qf, bk1, s1);
    }

    const bool vk0 = vkp[kt * 32 + n] != 0;
    const bool vk1 = vkp[kt * 32 + 16 + n] != 0;

    half_t* Pw = &sP[wid][0];
#pragma unroll
    for (int r = 0; r < 8; ++r) {
      const float a0 = (vqb[r] && vk0) ? s0[r] : -1e9f;
      const float a1 = (vqb[r] && vk1) ? s1[r] : -1e9f;
      float mx = fmaxf(a0, a1);
#pragma unroll
      for (int o = 8; o > 0; o >>= 1) mx = fmaxf(mx, __shfl_xor(mx, o, 32));
      const float mnew = fmaxf(m_run[r], mx);
      const float alpha = __expf(m_run[r] - mnew);
      const float p0 = __expf(a0 - mnew);
      const float p1 = __expf(a1 - mnew);
      float rs = p0 + p1;
#pragma unroll
      for (int o = 8; o > 0; o >>= 1) rs += __shfl_xor(rs, o, 32);
      l_run[r] = l_run[r] * alpha + rs;
      m_run[r] = mnew;
      o0[r] *= alpha;
      o1[r] *= alpha;
      Pw[(hl * 8 + r) * LDS_T + n] = (half_t)p0;
      Pw[(hl * 8 + r) * LDS_T + 16 + n] = (half_t)p1;
    }
    // same-wave LDS RAW: make P writes visible before fragment reads
    asm volatile("s_wait_dscnt 0" ::: "memory");
    {
      v16h pf = load_a_frag_vec(Pw, LDS_T);
      v16h bv0 = load_bt_frag_vec(sVt, LDS_T);
      v16h bv1 = load_bt_frag_vec(sVt + 16 * LDS_T, LDS_T);
      o0 = WMMA_F32_F16(pf, bv0, o0);
      o1 = WMMA_F32_F16(pf, bv1, o1);
    }
    __syncthreads();
  }

#pragma unroll
  for (int sub = 0; sub < 2; ++sub) {
    const v8f& oo = sub ? o1 : o0;
#pragma unroll
    for (int r = 0; r < 8; ++r) {
      const int l = qbase + hl * 8 + r;
      const int dk = sub * 16 + n;
      Om[((size_t)(bb * L_) + l) * D_ + h * DK_ + dk] = (half_t)(oo[r] / l_run[r]);
    }
  }
}

// ---------------- output projection + bias + residual ----------------
__global__ __launch_bounds__(256) void oproj_kernel(
    const half_t* __restrict__ OA, const half_t* __restrict__ OB,
    const half_t* __restrict__ WOT, const float* __restrict__ bo,
    const float* __restrict__ xa, const float* __restrict__ xb,
    float* __restrict__ TA, float* __restrict__ TB) {
  __shared__ alignas(16) half_t sA[128 * LDS_T];
  __shared__ alignas(16) half_t sBt[64 * LDS_T];
  const int z = blockIdx.z;
  const half_t* O = z ? OB : OA;
  const float* xr = z ? xb : xa;
  float* T = z ? TB : TA;
  const int wid = threadIdx.x >> 5, lane = threadIdx.x & 31;
  const int mpair = wid >> 1, npair = wid & 1;
  const int n = lane & 15, hl = lane >> 4;
  v8f acc[2][2] = {};
  gemm_mainloop(O + (size_t)blockIdx.x * 128 * D_, D_,
                WOT + (size_t)blockIdx.y * 64 * D_, D_, D_,
                sA, sBt, acc, mpair, npair);
  const int rowBase = blockIdx.x * 128 + mpair * 32 + hl * 8;
  const int colBase = blockIdx.y * 64 + npair * 32;
#pragma unroll
  for (int mi = 0; mi < 2; ++mi) {
#pragma unroll
    for (int ni = 0; ni < 2; ++ni) {
      const v8f& a = acc[mi][ni];
      const int col = colBase + ni * 16 + n;
      const float bi = bo[col];
#pragma unroll
      for (int r = 0; r < 8; ++r) {
        const size_t idx = (size_t)(rowBase + mi * 16 + r) * D_ + col;
        T[idx] = a[r] + bi + xr[idx];
      }
    }
  }
}

// ---------------- mid double masked LN (t in-place f32, xn f16) ----------------
__global__ __launch_bounds__(256) void ln_mid_kernel(
    float* __restrict__ TA, float* __restrict__ TB,
    const unsigned char* __restrict__ va, const unsigned char* __restrict__ vb,
    const float* __restrict__ goa, const float* __restrict__ boa,
    const float* __restrict__ gob, const float* __restrict__ bob,
    const float* __restrict__ fg, const float* __restrict__ fb,
    half_t* __restrict__ XNA, half_t* __restrict__ XNB) {
  const int z = blockIdx.y;
  float* T = z ? TB : TA;
  const unsigned char* vv = z ? vb : va;
  const float* g1 = z ? gob : goa;
  const float* b1 = z ? bob : boa;
  half_t* XN = z ? XNB : XNA;
  const int wid = threadIdx.x >> 5, lane = threadIdx.x & 31;
  const size_t row = (size_t)blockIdx.x * 8 + wid;
  float* rp = T + row * D_;
  const bool valid = vv[row] != 0;
  float x[6]; float s = 0.f, ss = 0.f;
#pragma unroll
  for (int i = 0; i < 6; ++i) { x[i] = rp[lane + 32 * i]; s += x[i]; ss += x[i] * x[i]; }
  s = wave_sum(s); ss = wave_sum(ss);
  const float mu = s * (1.f / D_);
  const float rstd = rsqrtf(ss * (1.f / D_) - mu * mu + 1e-5f);
  float t[6]; float s2 = 0.f, ss2 = 0.f;
#pragma unroll
  for (int i = 0; i < 6; ++i) {
    const int c = lane + 32 * i;
    t[i] = valid ? (x[i] - mu) * rstd * g1[c] + b1[c] : x[i];
    s2 += t[i]; ss2 += t[i] * t[i];
  }
#pragma unroll
  for (int i = 0; i < 6; ++i) rp[lane + 32 * i] = t[i];
  s2 = wave_sum(s2); ss2 = wave_sum(ss2);
  const float mu2 = s2 * (1.f / D_);
  const float rstd2 = rsqrtf(ss2 * (1.f / D_) - mu2 * mu2 + 1e-5f);
  half_t* xp = XN + row * D_;
#pragma unroll
  for (int i = 0; i < 6; ++i) {
    const int c = lane + 32 * i;
    const float y = valid ? (t[i] - mu2) * rstd2 * fg[c] + fb[c] : t[i];
    xp[c] = (half_t)y;
  }
}

// ---------------- FFN GEMM1 + exact GELU ----------------
__global__ __launch_bounds__(256) void ffn1_kernel(
    const half_t* __restrict__ XNA, const half_t* __restrict__ XNB,
    const half_t* __restrict__ W1T, const float* __restrict__ b1,
    half_t* __restrict__ H1A, half_t* __restrict__ H1B) {
  __shared__ alignas(16) half_t sA[128 * LDS_T];
  __shared__ alignas(16) half_t sBt[64 * LDS_T];
  const int z = blockIdx.z;
  const half_t* X = z ? XNB : XNA;
  half_t* Hh = z ? H1B : H1A;
  const int wid = threadIdx.x >> 5, lane = threadIdx.x & 31;
  const int mpair = wid >> 1, npair = wid & 1;
  const int n = lane & 15, hl = lane >> 4;
  v8f acc[2][2] = {};
  gemm_mainloop(X + (size_t)blockIdx.x * 128 * D_, D_,
                W1T + (size_t)blockIdx.y * 64 * D_, D_, D_,
                sA, sBt, acc, mpair, npair);
  const int rowBase = blockIdx.x * 128 + mpair * 32 + hl * 8;
  const int colBase = blockIdx.y * 64 + npair * 32;
#pragma unroll
  for (int mi = 0; mi < 2; ++mi) {
#pragma unroll
    for (int ni = 0; ni < 2; ++ni) {
      const v8f& a = acc[mi][ni];
      const int col = colBase + ni * 16 + n;
      const float bi = b1[col];
#pragma unroll
      for (int r = 0; r < 8; ++r) {
        const float v = a[r] + bi;
        const float gel = 0.5f * v * (1.f + erff(v * 0.70710678118654752f));
        Hh[(size_t)(rowBase + mi * 16 + r) * HID_ + col] = (half_t)gel;
      }
    }
  }
}

// ---------------- FFN GEMM2 + bias + residual ----------------
__global__ __launch_bounds__(256) void ffn2_kernel(
    const half_t* __restrict__ H1A, const half_t* __restrict__ H1B,
    const half_t* __restrict__ W2T, const float* __restrict__ b2,
    const float* __restrict__ TA, const float* __restrict__ TB,
    float* __restrict__ UA, float* __restrict__ UB) {
  __shared__ alignas(16) half_t sA[128 * LDS_T];
  __shared__ alignas(16) half_t sBt[64 * LDS_T];
  const int z = blockIdx.z;
  const half_t* Hh = z ? H1B : H1A;
  const float* T = z ? TB : TA;
  float* U = z ? UB : UA;
  const int wid = threadIdx.x >> 5, lane = threadIdx.x & 31;
  const int mpair = wid >> 1, npair = wid & 1;
  const int n = lane & 15, hl = lane >> 4;
  v8f acc[2][2] = {};
  gemm_mainloop(Hh + (size_t)blockIdx.x * 128 * HID_, HID_,
                W2T + (size_t)blockIdx.y * 64 * HID_, HID_, HID_,
                sA, sBt, acc, mpair, npair);
  const int rowBase = blockIdx.x * 128 + mpair * 32 + hl * 8;
  const int colBase = blockIdx.y * 64 + npair * 32;
#pragma unroll
  for (int mi = 0; mi < 2; ++mi) {
#pragma unroll
    for (int ni = 0; ni < 2; ++ni) {
      const v8f& a = acc[mi][ni];
      const int col = colBase + ni * 16 + n;
      const float bi = b2[col];
#pragma unroll
      for (int r = 0; r < 8; ++r) {
        const size_t idx = (size_t)(rowBase + mi * 16 + r) * D_ + col;
        U[idx] = a[r] + bi + T[idx];
      }
    }
  }
}

// ---------------- final masked LN -> d_out ----------------
__global__ __launch_bounds__(256) void ln_out_kernel(
    const float* __restrict__ UA, const float* __restrict__ UB,
    const unsigned char* __restrict__ va, const unsigned char* __restrict__ vb,
    const float* __restrict__ g, const float* __restrict__ be,
    float* __restrict__ dout) {
  const int z = blockIdx.y;
  const float* U = z ? UB : UA;
  const unsigned char* vv = z ? vb : va;
  float* out = dout + (size_t)z * M_ * D_;
  const int wid = threadIdx.x >> 5, lane = threadIdx.x & 31;
  const size_t row = (size_t)blockIdx.x * 8 + wid;
  const float* rp = U + row * D_;
  float x[6]; float s = 0.f, ss = 0.f;
#pragma unroll
  for (int i = 0; i < 6; ++i) { x[i] = rp[lane + 32 * i]; s += x[i]; ss += x[i] * x[i]; }
  s = wave_sum(s); ss = wave_sum(ss);
  const float mu = s * (1.f / D_);
  const float rstd = rsqrtf(ss * (1.f / D_) - mu * mu + 1e-5f);
  const bool valid = vv[row] != 0;
  float* op = out + row * D_;
#pragma unroll
  for (int i = 0; i < 6; ++i) {
    const int c = lane + 32 * i;
    op[c] = valid ? (x[i] - mu) * rstd * g[c] + be[c] : x[i];
  }
}

// ---------------- host launcher ----------------
extern "C" void kernel_launch(void* const* d_in, const int* in_sizes, int n_in,
                              void* d_out, int out_size, void* d_ws, size_t ws_size,
                              hipStream_t stream) {
  (void)in_sizes; (void)n_in; (void)out_size; (void)ws_size;
  const float* x_a     = (const float*)d_in[0];
  const float* x_b     = (const float*)d_in[1];
  const unsigned char* valid_a = (const unsigned char*)d_in[2];  // jax bool = 1 byte
  const unsigned char* valid_b = (const unsigned char*)d_in[3];
  const float* ln_a_g  = (const float*)d_in[4];
  const float* ln_a_b  = (const float*)d_in[5];
  const float* ln_b_g  = (const float*)d_in[6];
  const float* ln_b_b  = (const float*)d_in[7];
  const float* ln_oa_g = (const float*)d_in[8];
  const float* ln_oa_b = (const float*)d_in[9];
  const float* ln_ob_g = (const float*)d_in[10];
  const float* ln_ob_b = (const float*)d_in[11];
  const float* wq = (const float*)d_in[12];
  const float* bq = (const float*)d_in[13];
  const float* wk = (const float*)d_in[14];
  const float* bk = (const float*)d_in[15];
  const float* wv = (const float*)d_in[16];
  const float* bv = (const float*)d_in[17];
  const float* wo = (const float*)d_in[18];
  const float* bo = (const float*)d_in[19];
  const float* fln_g  = (const float*)d_in[20];
  const float* fln_b  = (const float*)d_in[21];
  const float* flno_g = (const float*)d_in[22];
  const float* flno_b = (const float*)d_in[23];
  const float* w1 = (const float*)d_in[24];
  const float* b1 = (const float*)d_in[25];
  const float* w2 = (const float*)d_in[26];
  const float* b2 = (const float*)d_in[27];

  char* ws = (char*)d_ws;
  size_t off = 0;
  auto alloc = [&](size_t bytes) -> char* {
    char* p = ws + off;
    off = (off + bytes + 255) & ~(size_t)255;
    return p;
  };

  const size_t S  = (size_t)D_ * D_;    // 36864
  const size_t SW = (size_t)D_ * HID_;  // 147456
  const size_t SLOT = (size_t)M_ * D_;  // halves per [B,L,D]-f16 tensor

  half_t* wpool = (half_t*)alloc((4 * S + 2 * SW) * sizeof(half_t));
  half_t* WQT = wpool;
  half_t* WKT = wpool + S;
  half_t* WVT = wpool + 2 * S;
  half_t* WOT = wpool + 3 * S;
  half_t* W1T = wpool + 4 * S;
  half_t* W2T = wpool + 4 * S + SW;

  half_t* A16  = (half_t*)alloc(SLOT * sizeof(half_t));   // also xn_a after ln_mid
  half_t* B16h = (half_t*)alloc(SLOT * sizeof(half_t));   // also xn_b after ln_mid
  half_t* QKV  = (half_t*)alloc(8 * SLOT * sizeof(half_t));
  half_t* QA  = QKV + 0 * SLOT;
  half_t* KA  = QKV + 1 * SLOT;
  half_t* VAT = QKV + 2 * SLOT;
  half_t* QB  = QKV + 3 * SLOT;
  half_t* KB  = QKV + 4 * SLOT;
  half_t* VBT = QKV + 5 * SLOT;
  half_t* OA16 = QKV + 6 * SLOT;
  half_t* OB16 = QKV + 7 * SLOT;
  half_t* H1A = QKV;             // reuses slots 0-3 after attention is done
  half_t* H1B = QKV + 4 * SLOT;  // reuses slots 4-7 after oproj is done

  float* T0A = (float*)alloc(SLOT * sizeof(float));  // t0 then t (in-place LN)
  float* T0B = (float*)alloc(SLOT * sizeof(float));
  float* U0A = (float*)alloc(SLOT * sizeof(float));
  float* U0B = (float*)alloc(SLOT * sizeof(float));

  const int cvtN = (int)(4 * S + 2 * SW);
  cvt_w_kernel<<<dim3((cvtN + 255) / 256), 256, 0, stream>>>(wq, wk, wv, wo, w1, w2, wpool);

  ln_in_kernel<<<dim3(M_ / 8, 2), 256, 0, stream>>>(
      x_a, x_b, valid_a, valid_b, ln_a_g, ln_a_b, ln_b_g, ln_b_b, A16, B16h);

  proj_kernel<<<dim3(M_ / 128, D_ / 64, 6), 256, 0, stream>>>(
      A16, B16h, WQT, WKT, WVT, bq, bk, bv, QA, KA, VAT, QB, KB, VBT);

  attn_kernel<<<dim3(B_ * H_ * (L_ / 128), 2), 256, 0, stream>>>(
      QA, KB, VBT, QB, KA, VAT, valid_a, valid_b, OA16, OB16);

  oproj_kernel<<<dim3(M_ / 128, D_ / 64, 2), 256, 0, stream>>>(
      OA16, OB16, WOT, bo, x_a, x_b, T0A, T0B);

  ln_mid_kernel<<<dim3(M_ / 8, 2), 256, 0, stream>>>(
      T0A, T0B, valid_a, valid_b, ln_oa_g, ln_oa_b, ln_ob_g, ln_ob_b,
      fln_g, fln_b, A16, B16h);

  ffn1_kernel<<<dim3(M_ / 128, HID_ / 64, 2), 256, 0, stream>>>(
      A16, B16h, W1T, b1, H1A, H1B);

  ffn2_kernel<<<dim3(M_ / 128, D_ / 64, 2), 256, 0, stream>>>(
      H1A, H1B, W2T, b2, T0A, T0B, U0A, U0B);

  ln_out_kernel<<<dim3(M_ / 8, 2), 256, 0, stream>>>(
      U0A, U0B, valid_a, valid_b, flno_g, flno_b, (float*)d_out);
}
